// BertSelfAttention_32521492365988
// MI455X (gfx1250) — matrix operand order repaired and checked by reading, not compiled, not run on hardware
//
#include <hip/hip_runtime.h>

// ---------------- problem constants ----------------
constexpr int B_   = 4;
constexpr int S_   = 2048;
constexpr int HID_ = 768;
constexpr int H_   = 12;
constexpr int DBLK_= 256;   // HID/3
constexpr int HD_  = 64;    // head dim

typedef __attribute__((ext_vector_type(16))) _Float16 v16h;
typedef __attribute__((ext_vector_type(8)))  _Float16 v8h;
typedef __attribute__((ext_vector_type(8)))  float    v8f;

static __device__ __forceinline__ v16h jn(const _Float16* lo, const _Float16* hi) {
    v8h a = *(const v8h*)lo;
    v8h b = *(const v8h*)hi;
    v16h r;
#pragma unroll
    for (int i = 0; i < 8; ++i) { r[i] = a[i]; r[i + 8] = b[i]; }
    return r;
}

// =====================================================================
// Kernel 1: block-diagonal QKV projection.
//   x[B,S,768] (f32) -> Q[B,H,S,64] (f16, pre-scaled by log2e/8),
//                       K[B,H,S,64] (f16),
//                       Vt[B,H,64,S] (f16, transposed for PV B-fragments)
// grid = (S/64, H, B), block = 384 (12 waves: {Q,K,V} x 4 M-tiles)
// =====================================================================
__global__ __launch_bounds__(384) void qkv_proj_kernel(
    const float* __restrict__ x,
    const float* __restrict__ Wq, const float* __restrict__ bq,
    const float* __restrict__ Wk, const float* __restrict__ bk,
    const float* __restrict__ Wv, const float* __restrict__ bv,
    _Float16* __restrict__ qo, _Float16* __restrict__ ko, _Float16* __restrict__ vto)
{
    constexpr int LDA = 264;   // halfs per sA row (528B, 16B-aligned)
    constexpr int LDW = 48;    // halfs per sW row (96B, 16B-aligned)
    __shared__ alignas(16) _Float16 sA[64 * LDA];       // 33.8 KB: x tile 64x256 f16
    __shared__ alignas(16) _Float16 sW[3 * 64 * LDW];   // 18 KB: W k-slab, transposed [mat][e][kk]

    const int s0  = blockIdx.x * 64;
    const int h   = blockIdx.y;
    const int b   = blockIdx.z;
    const int m   = h >> 2;              // modality (hpm = 4)
    const int tid = threadIdx.x;

    // ---- stage x tile (64 x 256 f32 -> f16) ----
    const float* xblk = x + ((size_t)b * S_ + s0) * HID_ + (size_t)m * DBLK_;
    for (int i = tid; i < 64 * 64; i += 384) {       // float4 granularity
        const int r  = i >> 6;
        const int c4 = i & 63;
        const float4 f = *(const float4*)(xblk + (size_t)r * HID_ + c4 * 4);
        _Float16* dst = &sA[r * LDA + c4 * 4];
        dst[0] = (_Float16)f.x; dst[1] = (_Float16)f.y;
        dst[2] = (_Float16)f.z; dst[3] = (_Float16)f.w;
    }

    const int wave = tid >> 5;      // 0..11
    const int lane = tid & 31;
    const int mat  = wave >> 2;     // 0=Q 1=K 2=V
    const int mt   = wave & 3;      // M-tile within 64 rows
    const int l15  = lane & 15;
    const int akb  = (lane >> 4) * 8;    // A-frag K sub-base
    const int bkb  = (lane >> 4) * 8;    // B-frag K sub-base
    const int rbase= (lane >> 4) * 8;    // C-frag row base

    const float* Wm[3] = { Wq + (size_t)h * DBLK_ * HD_,
                           Wk + (size_t)h * DBLK_ * HD_,
                           Wv + (size_t)h * DBLK_ * HD_ };

    v8f acc[4] = {};

    for (int k0 = 0; k0 < DBLK_; k0 += 32) {
        __syncthreads();   // protect sW WAR across iterations
        // stage W k-slab: 3 mats x 32 x 64 f32 -> f16, transposed -> sW[mat][e][kk]
        for (int i = tid; i < 3 * 32 * 64; i += 384) {
            const int mm  = i >> 11;          // /2048
            const int rem = i & 2047;
            const int kk  = rem >> 6;
            const int e   = rem & 63;
            sW[(mm * 64 + e) * LDW + kk] = (_Float16)Wm[mm][(size_t)(k0 + kk) * HD_ + e];
        }
        __syncthreads();

        // A fragment: 16 rows (mt tile) x 32 K
        const _Float16* ap = &sA[(mt * 16 + l15) * LDA + k0 + akb];
        const v16h a = jn(ap, ap + 16);
#pragma unroll
        for (int nt = 0; nt < 4; ++nt) {
            const _Float16* bp = &sW[(mat * 64 + nt * 16 + l15) * LDW + bkb];
            const v16h bf = jn(bp, bp + 16);
            acc[nt] = __builtin_amdgcn_wmma_f32_16x16x32_f16(
                false, a, false, bf, (short)0, acc[nt], false, false);
        }
    }

    // ---- epilogue: bias (+ scale into Q), store f16 ----
    const float* bias = (mat == 0 ? bq : (mat == 1 ? bk : bv)) + (size_t)h * HD_;
    // fold 1/sqrt(64) and log2(e) into Q so softmax can use exp2 (v_exp_f32 is base-2)
    const float qscale = (mat == 0) ? 0.18033688011112042f : 1.0f;

#pragma unroll
    for (int nt = 0; nt < 4; ++nt) {
        const int e  = nt * 16 + l15;
        const float be = bias[e];
        if (mat == 2) {
            // Vt[b][h][e][s] ; 8 consecutive s per lane -> one 16B store
            v8h pk;
#pragma unroll
            for (int r = 0; r < 8; ++r) pk[r] = (_Float16)(acc[nt][r] + be);
            _Float16* dst = vto + (((size_t)b * H_ + h) * HD_ + e) * S_
                                + s0 + mt * 16 + rbase;
            *(v8h*)dst = pk;
        } else {
            _Float16* base = (mat == 0 ? qo : ko)
                + (((size_t)b * H_ + h) * S_ + s0 + mt * 16 + rbase) * HD_ + e;
#pragma unroll
            for (int r = 0; r < 8; ++r)
                base[(size_t)r * HD_] = (_Float16)((acc[nt][r] + be) * qscale);
        }
    }
}

// =====================================================================
// Kernel 2: flash attention with double-buffered async K/V staging.
//   out[b][s][h*64+e] = softmax2(Q Kt) V   (Q pre-scaled by log2e/8)
// grid = (S/64, H, B), block = 128 (4 waves, 16 query rows each)
// =====================================================================
constexpr int LDK = 72;   // halfs per staged row (144B, 16B-aligned)

// Issue 128 rows x 64B of K and of Vt into LDS via async DMA.
// 512 x 16B chunks per tile, 128 threads -> 4 K + 4 V instr per thread
// (ASYNCcnt += 8 per wave).
static __device__ __forceinline__ void issue_kv_async(
    const _Float16* kb, const _Float16* vb, int t0,
    unsigned ldsK, unsigned ldsV, int tid)
{
#pragma unroll
    for (int j = 0; j < 4; ++j) {
        const int c   = j * 128 + tid;
        const int row = c >> 2;          // 0..127
        const int c16 = c & 3;           // 16B chunk within first 64B of row
        const unsigned kOff = (unsigned)((t0 + row) * (HD_ * 2) + c16 * 16);
        const unsigned kDst = ldsK + (unsigned)(row * (LDK * 2) + c16 * 16);
        asm volatile("global_load_async_to_lds_b128 %0, %1, %2"
                     :: "v"(kDst), "v"(kOff), "s"(kb) : "memory");
        const unsigned vOff = (unsigned)((row * S_ + t0) * 2 + c16 * 16);
        const unsigned vDst = ldsV + (unsigned)(row * (LDK * 2) + c16 * 16);
        asm volatile("global_load_async_to_lds_b128 %0, %1, %2"
                     :: "v"(vDst), "v"(vOff), "s"(vb) : "memory");
    }
}

__global__ __launch_bounds__(128) void flash_attn_kernel(
    const _Float16* __restrict__ q,
    const _Float16* __restrict__ k,
    const _Float16* __restrict__ vt,
    float* __restrict__ out)
{
    __shared__ alignas(16) _Float16 sK[2][64 * LDK];   // 2 x 9.2 KB
    __shared__ alignas(16) _Float16 sV[2][64 * LDK];   // 2 x 9.2 KB
    __shared__ alignas(16) _Float16 sP[4][16 * LDK];   // per-wave probs staging

    const int tid  = threadIdx.x;
    const int h    = blockIdx.y;
    const int b    = blockIdx.z;
    const int wave = tid >> 5;
    const int lane = tid & 31;
    const int l15  = lane & 15;
    const int akb  = (lane >> 4) * 8;
    const int bkb  = (lane >> 4) * 8;
    const int rbase= (lane >> 4) * 8;
    const int srow0= blockIdx.x * 64 + wave * 16;

    const _Float16* qb = q  + ((size_t)b * H_ + h) * S_ * HD_;
    const _Float16* kb = k  + ((size_t)b * H_ + h) * S_ * HD_;
    const _Float16* vb = vt + ((size_t)b * H_ + h) * HD_ * S_;

    const unsigned ldsK0 = (unsigned)(uintptr_t)&sK[0][0];
    const unsigned ldsK1 = (unsigned)(uintptr_t)&sK[1][0];
    const unsigned ldsV0 = (unsigned)(uintptr_t)&sV[0][0];
    const unsigned ldsV1 = (unsigned)(uintptr_t)&sV[1][0];

    // Q A-fragments (16 rows x HD=64 -> two K-steps)
    v16h qf[2];
#pragma unroll
    for (int kk = 0; kk < 2; ++kk) {
        const _Float16* p = qb + (size_t)(srow0 + l15) * HD_ + kk * 32 + akb;
        qf[kk] = jn(p, p + 16);
    }

    float mrow[8], lrow[8];
    v8f accO[4] = {};
#pragma unroll
    for (int r = 0; r < 8; ++r) { mrow[r] = -1e30f; lrow[r] = 0.f; }

    _Float16* ps = sP[wave];

    // prologue: prefetch tile 0 into buffer 0
    issue_kv_async(kb, vb, 0, ldsK0, ldsV0, tid);

    for (int t0 = 0; t0 < S_; t0 += 64) {
        const int buf = (t0 >> 6) & 1;
        const _Float16* cK = sK[buf];
        const _Float16* cV = sV[buf];

        if (t0 + 64 < S_) {
            // prefetch next tile into the other buffer, then wait for current
            issue_kv_async(kb, vb, t0 + 64,
                           buf ? ldsK0 : ldsK1, buf ? ldsV0 : ldsV1, tid);
            asm volatile("s_wait_asynccnt 0x8" ::: "memory");
        } else {
            asm volatile("s_wait_asynccnt 0x0" ::: "memory");
        }
        __syncthreads();   // current tile visible to all waves

        // ---- scores: 16 x 64 tile ----
        v8f sc[4] = {};
#pragma unroll
        for (int nt = 0; nt < 4; ++nt) {
#pragma unroll
            for (int kk = 0; kk < 2; ++kk) {
                const _Float16* p = cK + (nt * 16 + l15) * LDK + kk * 32 + bkb;
                const v16h kf = jn(p, p + 16);
                sc[nt] = __builtin_amdgcn_wmma_f32_16x16x32_f16(
                    false, qf[kk], false, kf, (short)0, sc[nt], false, false);
            }
        }

        // ---- online softmax in base-2 (rows live in 16-lane halves) ----
        float tmax[8];
#pragma unroll
        for (int r = 0; r < 8; ++r) {
            float v = sc[0][r];
            v = fmaxf(v, sc[1][r]); v = fmaxf(v, sc[2][r]); v = fmaxf(v, sc[3][r]);
            tmax[r] = v;
        }
#pragma unroll
        for (int off = 1; off < 16; off <<= 1)
#pragma unroll
            for (int r = 0; r < 8; ++r)
                tmax[r] = fmaxf(tmax[r], __shfl_xor(tmax[r], off, 32));

        float alpha[8];
#pragma unroll
        for (int r = 0; r < 8; ++r) {
            const float mn = fmaxf(mrow[r], tmax[r]);
            alpha[r] = __builtin_amdgcn_exp2f(mrow[r] - mn);
            mrow[r]  = mn;
        }

        float rs[8];
#pragma unroll
        for (int r = 0; r < 8; ++r) rs[r] = 0.f;
#pragma unroll
        for (int nt = 0; nt < 4; ++nt)
#pragma unroll
            for (int r = 0; r < 8; ++r) {
                const float p = __builtin_amdgcn_exp2f(sc[nt][r] - mrow[r]);
                sc[nt][r] = p;
                rs[r] += p;
            }
#pragma unroll
        for (int off = 1; off < 16; off <<= 1)
#pragma unroll
            for (int r = 0; r < 8; ++r)
                rs[r] += __shfl_xor(rs[r], off, 32);
#pragma unroll
        for (int r = 0; r < 8; ++r) lrow[r] = lrow[r] * alpha[r] + rs[r];
#pragma unroll
        for (int nt = 0; nt < 4; ++nt)
#pragma unroll
            for (int r = 0; r < 8; ++r) accO[nt][r] *= alpha[r];

        // ---- re-layout probs C-frag -> A-frag via per-wave LDS ----
        // sP is private per wave; LDS ops are in-order per wave (DScnt),
        // so a wave-level scheduling barrier is sufficient.
#pragma unroll
        for (int nt = 0; nt < 4; ++nt)
#pragma unroll
            for (int r = 0; r < 8; ++r)
                ps[(rbase + r) * LDK + nt * 16 + l15] = (_Float16)sc[nt][r];
        __builtin_amdgcn_wave_barrier();

        // ---- PV: accO += P @ V ----
#pragma unroll
        for (int kk = 0; kk < 2; ++kk) {
            const _Float16* pp = ps + l15 * LDK + kk * 32 + akb;
            const v16h pf = jn(pp, pp + 16);
#pragma unroll
            for (int nt = 0; nt < 4; ++nt) {
                const _Float16* vp = cV + (nt * 16 + l15) * LDK + kk * 32 + bkb;
                const v16h vf = jn(vp, vp + 16);
                accO[nt] = __builtin_amdgcn_wmma_f32_16x16x32_f16(
                    false, pf, false, vf, (short)0, accO[nt], false, false);
            }
        }
        __syncthreads();   // done reading this buffer before it is re-filled
    }

    // ---- normalize + store: out[b][s][h*64+e] ----
    float* ob = out + (size_t)b * S_ * HID_ + (size_t)h * HD_;
#pragma unroll
    for (int nt = 0; nt < 4; ++nt) {
        const int e = nt * 16 + l15;
#pragma unroll
        for (int r = 0; r < 8; ++r) {
            const int s = srow0 + rbase + r;
            ob[(size_t)s * HID_ + e] = accO[nt][r] / lrow[r];
        }
    }
}

// =====================================================================
extern "C" void kernel_launch(void* const* d_in, const int* in_sizes, int n_in,
                              void* d_out, int out_size, void* d_ws, size_t ws_size,
                              hipStream_t stream) {
    const float* x  = (const float*)d_in[0];
    const float* Wq = (const float*)d_in[1];
    const float* bq = (const float*)d_in[2];
    const float* Wk = (const float*)d_in[3];
    const float* bk = (const float*)d_in[4];
    const float* Wv = (const float*)d_in[5];
    const float* bv = (const float*)d_in[6];
    float* out = (float*)d_out;

    const size_t per = (size_t)B_ * H_ * S_ * HD_;   // 6,291,456 f16 elements
    _Float16* qws = (_Float16*)d_ws;
    _Float16* kws = qws + per;
    _Float16* vws = kws + per;                        // total 37.75 MB of ws

    dim3 g(S_ / 64, H_, B_);
    qkv_proj_kernel<<<g, 384, 0, stream>>>(x, Wq, bq, Wk, bk, Wv, bv, qws, kws, vws);
    flash_attn_kernel<<<g, 128, 0, stream>>>(qws, kws, vws, out);
}
